// GnnModel_23158463660653
// MI455X (gfx1250) — compile-verified, matrix operand order checked by reading, not run on hardware
//
#include <hip/hip_runtime.h>
#include <hip/hip_bf16.h>

typedef __attribute__((ext_vector_type(16))) __bf16 v16bf;
typedef __attribute__((ext_vector_type(8)))  float  v8f;

#define NN   20000
#define EE   320000
#define DD   256
#define EDIM 128
#define NL   5
#define BN_EPS 1e-5f

// ---- WMMA fragment helpers (wave32, ISA 7.12.2 layouts) --------------------
// A (16x32 bf16): lane&15 = row; lanes<16 hold K={0..7,16..23}, lanes>=16 K={8..15,24..31}
// B (32x16 bf16): stored column-major in LDS (Blds[n*stride + k]); same K split with lane&15 = col
__device__ __forceinline__ v16bf ld_frag(const __bf16* base, int stride) {
  const int lane = threadIdx.x & 31;
  const int rm   = lane & 15;
  const int kb   = (lane & 16) ? 8 : 0;
  const __bf16* p = base + rm * stride + kb;
  v16bf f;
#pragma unroll
  for (int i = 0; i < 8; ++i) { f[i] = p[i]; f[i + 8] = p[i + 16]; }
  return f;
}

__device__ __forceinline__ v8f wmma_bf16(v16bf a, v16bf b, v8f c) {
  return __builtin_amdgcn_wmma_f32_16x16x32_bf16(false, a, false, b, (short)0, c, false, false);
}

// ---- degree / norm precompute ----------------------------------------------
__global__ void k_setzero(float* p, int n) {
  int i = blockIdx.x * blockDim.x + threadIdx.x;
  if (i < n) p[i] = 0.0f;
}
__global__ void k_count(const int* __restrict__ rowi, float* __restrict__ deg) {
  int i = blockIdx.x * blockDim.x + threadIdx.x;   // grid covers EE exactly
  atomicAdd(&deg[rowi[i]], 1.0f);
}
__global__ void k_dinv(float* __restrict__ deg1, float* __restrict__ dinv, int n) {
  int i = blockIdx.x * blockDim.x + threadIdx.x;
  if (i < n) { float d = deg1[i] + 1.0f; deg1[i] = d; dinv[i] = rsqrtf(d); }
}
__global__ void k_norm(const int* __restrict__ rowi, const int* __restrict__ coli,
                       const float* __restrict__ dinv, float* __restrict__ nrm) {
  int i = blockIdx.x * blockDim.x + threadIdx.x;   // grid covers EE exactly
  nrm[i] = dinv[rowi[i]] * dinv[coli[i]];
}

// ---- per-layer zero (agg + BN stats) ---------------------------------------
__global__ void k_zero(float* __restrict__ agg, float* __restrict__ stats) {
  int i = blockIdx.x * blockDim.x + threadIdx.x;   // grid covers NN*DD/4 exactly
  float4 z; z.x = 0.f; z.y = 0.f; z.z = 0.f; z.w = 0.f;
  ((float4*)agg)[i] = z;
  if (blockIdx.x == 0 && threadIdx.x < 128) ((float4*)stats)[threadIdx.x] = z; // sums+sumsq (512 f32)
}

// ---- xl = h @ W[l] + b[l]  (M=20000, K=256, N=256) -------------------------
#define ASTR 34
#define BSTR 34
__global__ void k_xl(const float* __restrict__ h, const float* __restrict__ Wl,
                     const float* __restrict__ bl, float* __restrict__ xl) {
  __shared__ __bf16 Alds[16 * ASTR];
  __shared__ __bf16 Blds[256 * BSTR];
  const int tid  = threadIdx.x;          // 256 threads = 8 waves
  const int wave = tid >> 5;
  const int lane = tid & 31;
  const int mbase = blockIdx.x * 16;     // 1250 blocks, N=20000 divisible

  v8f acc0 = {}; v8f acc1 = {};
  for (int kt = 0; kt < 8; ++kt) {
    __syncthreads();
    { // stage A tile 16x32 (f32 -> bf16)
      int idx = tid * 2;
      int r = idx >> 5, c = idx & 31;
      const float* hp = h + (mbase + r) * DD + kt * 32 + c;
      Alds[r * ASTR + c]     = (__bf16)hp[0];
      Alds[r * ASTR + c + 1] = (__bf16)hp[1];
    }
    // stage B slab 32x256 transposed: Blds[n*BSTR + k] = W[k][n]
#pragma unroll 4
    for (int kk = 0; kk < 32; ++kk)
      Blds[tid * BSTR + kk] = (__bf16)Wl[(kt * 32 + kk) * DD + tid];
    __syncthreads();
    v16bf a  = ld_frag(Alds, ASTR);
    v16bf b0 = ld_frag(Blds + (wave * 2 + 0) * 16 * BSTR, BSTR);
    v16bf b1 = ld_frag(Blds + (wave * 2 + 1) * 16 * BSTR, BSTR);
    acc0 = wmma_bf16(a, b0, acc0);
    acc1 = wmma_bf16(a, b1, acc1);
  }
  const int n0 = (wave * 2 + 0) * 16 + (lane & 15);
  const int n1 = (wave * 2 + 1) * 16 + (lane & 15);
  const int mrow = (lane & 16) ? 8 : 0;
  const float bb0 = bl[n0], bb1 = bl[n1];
#pragma unroll
  for (int r = 0; r < 8; ++r) {
    int m = mbase + mrow + r;
    xl[m * DD + n0] = acc0[r] + bb0;
    xl[m * DD + n1] = acc1[r] + bb1;
  }
}

// ---- fused edge encoder GEMM + message + scatter-add -----------------------
// msg = norm * relu(xl[row] + edge_attr@We + be);  agg[col] += msg
#define WSTR   130
#define ESTR   130
#define CH_E   64
#define NCHUNK (EE / CH_E)   // 5000
__global__ void k_edge(const float* __restrict__ ea, const int* __restrict__ rowi,
                       const int* __restrict__ coli, const float* __restrict__ nrm,
                       const float* __restrict__ Wel, const float* __restrict__ bel,
                       const float* __restrict__ xl, float* __restrict__ agg) {
  __shared__ __bf16 WeLds[128 * WSTR];   // this block's 128-column half of We, col-major
  __shared__ __bf16 ELds[CH_E * ESTR];   // 64 edges x 128 feat, row-major
  __shared__ int    rowL[CH_E];
  __shared__ int    colL[CH_E];
  __shared__ float  nrmL[CH_E];

  const int tid   = threadIdx.x;         // 256 threads = 8 waves
  const int wave  = tid >> 5;
  const int lane  = tid & 31;
  const int half  = blockIdx.x & 1;      // which 128-column half of D this block owns
  const int nbase = half * 128;

  { // stage We half once per block: WeLds[n*WSTR + k] = We[k][nbase+n]
    int n  = tid & 127;
    int k0 = (tid >> 7) * 64;
    for (int k = 0; k < 64; ++k)
      WeLds[n * WSTR + k0 + k] = (__bf16)Wel[(k0 + k) * DD + nbase + n];
  }

  for (int ch = (blockIdx.x >> 1); ch < NCHUNK; ch += (gridDim.x >> 1)) {
    __syncthreads();                     // protect LDS reuse + first-pass We visibility
    const int ebase = ch * CH_E;
    if (tid < CH_E) {
      rowL[tid] = rowi[ebase + tid];
      colL[tid] = coli[ebase + tid];
      nrmL[tid] = nrm[ebase + tid];
    }
#pragma unroll 4
    for (int j = 0; j < 32; ++j) {       // 64x128 edge_attr tile, f32 -> bf16
      int idx = j * 256 + tid;
      int e = idx >> 7, k = idx & 127;
      ELds[e * ESTR + k] = (__bf16)ea[(ebase + e) * EDIM + k];
    }
    __syncthreads();

#pragma unroll
    for (int j = 0; j < 4; ++j) {        // 32 output tiles (4 edge-subtiles x 8 col-tiles)
      int combo = wave * 4 + j;
      int et = combo >> 3;               // 0..3
      int nt = combo & 7;                // 0..7
      v8f acc = {};
#pragma unroll
      for (int ks = 0; ks < 4; ++ks) {   // K = 128 in steps of 32
        v16bf a = ld_frag(ELds  + et * 16 * ESTR + ks * 32, ESTR);
        v16bf b = ld_frag(WeLds + nt * 16 * WSTR + ks * 32, WSTR);
        acc = wmma_bf16(a, b, acc);
      }
      int n = nbase + nt * 16 + (lane & 15);
      float bev = bel[n];
      int mrow = (lane & 16) ? 8 : 0;
#pragma unroll
      for (int r = 0; r < 8; ++r) {
        int el = et * 16 + mrow + r;     // edge within chunk
        float v = acc[r] + bev + xl[rowL[el] * DD + n];   // gather (L2-resident xl)
        v = v > 0.f ? v : 0.f;
        atomicAdd(&agg[colL[el] * DD + n], v * nrmL[el]); // scatter-add
      }
    }
  }
}

// ---- conv = agg + relu(xl + root)/deg, accumulate column stats -------------
__global__ void k_stats(const float* __restrict__ xl, float* __restrict__ conv,
                        const float* __restrict__ deg1, const float* __restrict__ rootl,
                        float* __restrict__ sums, float* __restrict__ sumsq) {
  const int c  = threadIdx.x;            // 256 columns
  const float rc = rootl[c];
  const int r0 = blockIdx.x * 100;       // 200 blocks x 100 rows
  float s = 0.f, q = 0.f;
  for (int i = 0; i < 100; ++i) {
    int row = r0 + i;
    float di = 1.0f / deg1[row];
    float x  = xl[row * DD + c] + rc;
    x = x > 0.f ? x : 0.f;
    float v = conv[row * DD + c] + x * di;
    conv[row * DD + c] = v;
    s += v; q += v * v;
  }
  atomicAdd(&sums[c], s);
  atomicAdd(&sumsq[c], q);
}

// ---- finalize BN affine ----------------------------------------------------
__global__ void k_bnfix(const float* __restrict__ sums, const float* __restrict__ sumsq,
                        const float* __restrict__ gml, const float* __restrict__ btl,
                        float* __restrict__ scale, float* __restrict__ shift) {
  int c = threadIdx.x;
  const float invN = 1.0f / (float)NN;
  float mean = sums[c] * invN;
  float var  = sumsq[c] * invN - mean * mean;   // biased var, matches jnp.var
  float inv  = rsqrtf(var + BN_EPS);
  float sc   = gml[c] * inv;
  scale[c] = sc;
  shift[c] = btl[c] - mean * sc;
}

// ---- h = act(conv*scale+shift) + h  (residual) -----------------------------
__global__ void k_apply(const float* __restrict__ conv, const float* __restrict__ scale,
                        const float* __restrict__ shift, const float* __restrict__ hin,
                        float* __restrict__ hout, int do_relu) {
  int i = blockIdx.x * blockDim.x + threadIdx.x;  // grid covers NN*DD/4 exactly
  int base = i * 4;
  int c = base & (DD - 1);
  float4 cv = ((const float4*)conv)[i];
  float4 hv = ((const float4*)hin)[i];
  float o0 = cv.x * scale[c]     + shift[c];
  float o1 = cv.y * scale[c + 1] + shift[c + 1];
  float o2 = cv.z * scale[c + 2] + shift[c + 2];
  float o3 = cv.w * scale[c + 3] + shift[c + 3];
  if (do_relu) {
    o0 = o0 > 0.f ? o0 : 0.f; o1 = o1 > 0.f ? o1 : 0.f;
    o2 = o2 > 0.f ? o2 : 0.f; o3 = o3 > 0.f ? o3 : 0.f;
  }
  float4 o; o.x = o0 + hv.x; o.y = o1 + hv.y; o.z = o2 + hv.z; o.w = o3 + hv.w;
  ((float4*)hout)[i] = o;
}

// ---------------------------------------------------------------------------
extern "C" void kernel_launch(void* const* d_in, const int* in_sizes, int n_in,
                              void* d_out, int out_size, void* d_ws, size_t ws_size,
                              hipStream_t stream) {
  (void)in_sizes; (void)n_in; (void)out_size; (void)ws_size;
  const float* input = (const float*)d_in[0];
  const int*   ei    = (const int*)  d_in[1];
  const float* ea    = (const float*)d_in[2];
  const float* W     = (const float*)d_in[3];
  const float* b     = (const float*)d_in[4];
  const float* root  = (const float*)d_in[5];
  const float* We    = (const float*)d_in[6];
  const float* be    = (const float*)d_in[7];
  const float* gm    = (const float*)d_in[8];
  const float* bt    = (const float*)d_in[9];
  float* out = (float*)d_out;
  const int* rowi = ei;
  const int* coli = ei + EE;

  float* ws    = (float*)d_ws;
  float* xl    = ws;                          // NN*DD
  float* agg   = xl  + (size_t)NN * DD;       // NN*DD (doubles as conv)
  float* nrm   = agg + (size_t)NN * DD;       // EE
  float* deg1  = nrm + EE;                    // NN
  float* dinv  = deg1 + NN;                   // NN
  float* sums  = dinv + NN;                   // DD
  float* sumsq = sums + DD;                   // DD
  float* scale = sumsq + DD;                  // DD
  float* shift = scale + DD;                  // DD

  // hoisted degree / symmetric norm
  k_setzero<<<(NN + 255) / 256, 256, 0, stream>>>(deg1, NN);
  k_count  <<<EE / 256,        256, 0, stream>>>(rowi, deg1);
  k_dinv   <<<(NN + 255) / 256, 256, 0, stream>>>(deg1, dinv, NN);
  k_norm   <<<EE / 256,        256, 0, stream>>>(rowi, coli, dinv, nrm);

  for (int l = 0; l < NL; ++l) {
    const float* hin   = (l == 0) ? input : out;
    const float* Wl    = W  + (size_t)l * DD * DD;
    const float* bl    = b  + l * DD;
    const float* rootl = root + l * DD;
    const float* Wel   = We + (size_t)l * EDIM * DD;
    const float* bel   = be + l * DD;
    const float* gml   = gm + l * DD;
    const float* btl   = bt + l * DD;

    k_zero <<<NN * DD / 1024, 256, 0, stream>>>(agg, sums);
    k_xl   <<<NN / 16,        256, 0, stream>>>(hin, Wl, bl, xl);
    k_edge <<<1000,           256, 0, stream>>>(ea, rowi, coli, nrm, Wel, bel, xl, agg);
    k_stats<<<200,            256, 0, stream>>>(xl, agg, deg1, rootl, sums, sumsq);
    k_bnfix<<<1,              DD,  0, stream>>>(sums, sumsq, gml, btl, scale, shift);
    k_apply<<<NN * DD / 1024, 256, 0, stream>>>(agg, scale, shift, hin, out, (l < NL - 1) ? 1 : 0);
  }
}
